// GSATViG_44590350467893
// MI455X (gfx1250) — matrix operand order repaired and checked by reading, not compiled, not run on hardware
//
#include <hip/hip_runtime.h>
#include <hip/hip_bf16.h>
#include <math.h>

typedef __attribute__((ext_vector_type(16))) _Float16 v16h;
typedef __attribute__((ext_vector_type(8)))  float    v8f;
typedef __attribute__((ext_vector_type(4)))  unsigned int v4u;

#define B_    64
#define IMG_  384
#define PATCH_ 16
#define C_    192
#define K_    5
#define N_    576          // 24*24 patches
#define BN_   (B_ * N_)    // 36864
#define HID_  768
#define E_    (BN_ * K_)   // 184320
#define PHID_ 1024

// ---------------- WMMA fragment helpers (wave32, 16x16x32 f16 -> f32) ----------------
// A 16x32 f16 fragment: lane L holds row M = L&15.
//   halves 0..7  = K[k0 + 8*(L>>4) + 0..7]
//   halves 8..15 = K[k0 + 16 + 8*(L>>4) + 0..7]
__device__ __forceinline__ v16h load_a_frag(const _Float16* row, int lane, int k0) {
  const _Float16* p = row + k0 + ((lane >> 4) << 3);
  v16h a;
  ((v4u*)&a)[0] = *(const v4u*)(p);        // ds_load_b128
  ((v4u*)&a)[1] = *(const v4u*)(p + 16);   // ds_load_b128
  return a;
}
// B 32x16 f16 fragment from an n-major weight Wt[n*K + k]: lane L holds column
// N = n0 + (L&15); halves 0..7 = K[k0 + 8*(L>>4) + 0..7], 8..15 = K[k0 + 16 + 8*(L>>4) + 0..7].
__device__ __forceinline__ v16h load_b_frag(const _Float16* Wt, int K, int lane,
                                            int k0, int n0) {
  const _Float16* p = Wt + (size_t)(n0 + (lane & 15)) * K + k0 + ((lane >> 4) << 3);
  v16h b;
  ((v4u*)&b)[0] = *(const v4u*)(p);        // global_load_b128
  ((v4u*)&b)[1] = *(const v4u*)(p + 16);   // global_load_b128
  return b;
}
__device__ __forceinline__ v8f wmma16(v16h a, v16h b, v8f c) {
  return __builtin_amdgcn_wmma_f32_16x16x32_f16(false, a, false, b, (short)0, c,
                                                false, false);
}
__device__ __forceinline__ float gelu_exact(float x) {
  return 0.5f * x * (1.0f + erff(x * 0.70710678118654752f));
}

// ---------------- weight down-convert kernels (once per launch, tiny) ----------------
__global__ void k_cvt(const float* __restrict__ s, _Float16* __restrict__ d, int n) {
  int i = blockIdx.x * blockDim.x + threadIdx.x;
  if (i < n) d[i] = (_Float16)s[i];
}
// transpose-convert: src is k-major [K][N] (x @ W convention) -> dst n-major [N][K]
__global__ void k_cvtT(const float* __restrict__ s, _Float16* __restrict__ d,
                       int Kd, int Nd) {
  int i = blockIdx.x * blockDim.x + threadIdx.x;
  if (i < Kd * Nd) {
    int n = i / Kd, k = i - n * Kd;
    d[i] = (_Float16)s[(size_t)k * Nd + n];
  }
}

// ---------------- stem: im2col + GEMM [16 patches x 768] x [768 x 192] ----------------
__global__ __launch_bounds__(128) void k_stem(
    const float* __restrict__ x, const _Float16* __restrict__ wstem,
    const float* __restrict__ stem_b, const float* __restrict__ pos,
    float* __restrict__ nf, _Float16* __restrict__ nf16) {
  __shared__ __align__(16) _Float16 As[16][776];   // 768 + pad, rows 16B aligned
  const int mtile = blockIdx.x;                    // 0..35
  const int b = blockIdx.y;
  const int tid = threadIdx.x, lane = tid & 31, wv = tid >> 5;
  for (int t = tid; t < 16 * 768; t += 128) {      // im2col stage
    int row = t / 768, k = t - row * 768;
    int pi = mtile * 16 + row;
    int hgp = pi / 24, wgp = pi - hgp * 24;
    int ci = k >> 8, py = (k >> 4) & 15, px = k & 15;
    As[row][k] = (_Float16)x[(((size_t)b * 3 + ci) * IMG_ + hgp * PATCH_ + py) * IMG_
                             + wgp * PATCH_ + px];
  }
  __syncthreads();
  const _Float16* arow = &As[lane & 15][0];
  for (int nt = wv * 3; nt < wv * 3 + 3; ++nt) {   // 12 N-tiles over 4 waves
    v8f acc = {};
#pragma unroll
    for (int ks = 0; ks < 24; ++ks) {
      v16h a = load_a_frag(arow, lane, ks * 32);
      v16h bb = load_b_frag(wstem, 768, lane, ks * 32, nt * 16);
      acc = wmma16(a, bb, acc);
    }
    int Nc = nt * 16 + (lane & 15);
    int mb = (lane >> 4) << 3;
    float sb = stem_b[Nc];
#pragma unroll
    for (int r = 0; r < 8; ++r) {
      int pi = mtile * 16 + r + mb;
      float v = acc[r] + sb + pos[Nc * N_ + pi];
      size_t o = ((size_t)b * N_ + pi) * C_ + Nc;
      nf[o] = v; nf16[o] = (_Float16)v;
    }
  }
}

__global__ void k_sq(const float* __restrict__ nf, float* __restrict__ sq) {
  int v = blockIdx.x * blockDim.x + threadIdx.x;
  if (v >= BN_) return;
  const float* p = nf + (size_t)v * C_;
  float s = 0.f;
  for (int c = 0; c < C_; ++c) s += p[c] * p[c];
  sq[v] = s;
}

// ---------------- kNN: Gram via WMMA + per-row top-5 ----------------
// ordering key per row n: sq[m] - 2*<h_n,h_m>  (sq[n] constant per row, dropped)
__global__ __launch_bounds__(128) void k_knn(
    const _Float16* __restrict__ h16, const float* __restrict__ sq,
    int* __restrict__ idxout) {
  __shared__ __align__(16) _Float16 As[16][200];
  __shared__ float drow[16][584];                  // 576 + pad
  const int mtile = blockIdx.x, b = blockIdx.y;
  const int tid = threadIdx.x, lane = tid & 31, wv = tid >> 5;
  for (int t = tid; t < 16 * C_; t += 128) {
    int row = t / C_, k = t - row * C_;
    As[row][k] = h16[((size_t)b * N_ + mtile * 16 + row) * C_ + k];
  }
  __syncthreads();
  const _Float16* arow = &As[lane & 15][0];
  const _Float16* bbase = h16 + (size_t)b * N_ * C_;  // node-major = n-major "weights"
  for (int nt = wv * 9; nt < wv * 9 + 9; ++nt) {      // 36 column tiles over 4 waves
    v8f acc = {};
#pragma unroll
    for (int ks = 0; ks < 6; ++ks) {
      v16h a = load_a_frag(arow, lane, ks * 32);
      v16h bb = load_b_frag(bbase, C_, lane, ks * 32, nt * 16);
      acc = wmma16(a, bb, acc);
    }
    int col = nt * 16 + (lane & 15);
    float sqm = sq[b * N_ + col];
    int mb = (lane >> 4) << 3;
#pragma unroll
    for (int r = 0; r < 8; ++r) {
      int M = r + mb;
      float d = sqm - 2.0f * acc[r];
      if (col == mtile * 16 + M) d = 1e9f;            // exclude self-loop
      drow[M][col] = d;
    }
  }
  __syncthreads();
  if (tid < 16) {
    int row = tid;
    float bd0 = 3e38f, bd1 = 3e38f, bd2 = 3e38f, bd3 = 3e38f, bd4 = 3e38f;
    int bi0 = 0, bi1 = 0, bi2 = 0, bi3 = 0, bi4 = 0;
    for (int m = 0; m < N_; ++m) {
      float d = drow[row][m];
      if (d < bd4) {
        if (d < bd0)      { bd4=bd3;bi4=bi3; bd3=bd2;bi3=bi2; bd2=bd1;bi2=bi1; bd1=bd0;bi1=bi0; bd0=d;bi0=m; }
        else if (d < bd1) { bd4=bd3;bi4=bi3; bd3=bd2;bi3=bi2; bd2=bd1;bi2=bi1; bd1=d;bi1=m; }
        else if (d < bd2) { bd4=bd3;bi4=bi3; bd3=bd2;bi3=bi2; bd2=d;bi2=m; }
        else if (d < bd3) { bd4=bd3;bi4=bi3; bd3=d;bi3=m; }
        else              { bd4=d;bi4=m; }
      }
    }
    int* o = idxout + ((size_t)b * N_ + mtile * 16 + row) * K_;
    o[0] = bi0; o[1] = bi1; o[2] = bi2; o[3] = bi3; o[4] = bi4;
  }
}

// ---------------- edge attention: fused [16E x 384]x[384x192] relu, dot w2, sigmoid ----
__global__ __launch_bounds__(32) void k_edge(
    const _Float16* __restrict__ h16, const int* __restrict__ idx,
    const _Float16* __restrict__ w1t, const float* __restrict__ b1,
    const float* __restrict__ w2, const float* __restrict__ b2,
    float* __restrict__ eatt) {
  __shared__ __align__(16) _Float16 As[16][392];     // 384 + pad
  const int e0 = blockIdx.x * 16;
  const int lane = threadIdx.x;
  for (int row = 0; row < 16; ++row) {
    int e = e0 + row;
    int vd = e / K_;
    int b = vd / N_;
    int src = b * N_ + idx[e];
    const _Float16* ps = h16 + (size_t)src * C_;
    const _Float16* pd = h16 + (size_t)vd * C_;
    for (int k = lane; k < C_; k += 32) { As[row][k] = ps[k]; As[row][C_ + k] = pd[k]; }
  }
  __syncthreads();
  float s[8] = {0.f, 0.f, 0.f, 0.f, 0.f, 0.f, 0.f, 0.f};
  const _Float16* arow = &As[lane & 15][0];
  for (int nt = 0; nt < 12; ++nt) {
    v8f acc = {};
#pragma unroll
    for (int ks = 0; ks < 12; ++ks) {
      v16h a = load_a_frag(arow, lane, ks * 32);
      v16h bb = load_b_frag(w1t, 384, lane, ks * 32, nt * 16);
      acc = wmma16(a, bb, acc);
    }
    int col = nt * 16 + (lane & 15);
    float bias = b1[col], wc = w2[col];
#pragma unroll
    for (int r = 0; r < 8; ++r) {
      float h = acc[r] + bias;
      s[r] += (h > 0.f ? h : 0.f) * wc;
    }
  }
#pragma unroll
  for (int r = 0; r < 8; ++r) {
#pragma unroll
    for (int off = 1; off < 16; off <<= 1) s[r] += __shfl_xor(s[r], off, 16);
  }
  if ((lane & 15) == 0) {
    int mb = (lane >> 4) << 3;
    float b2v = b2[0];
#pragma unroll
    for (int r = 0; r < 8; ++r) {
      float v = s[r] + b2v;
      eatt[e0 + mb + r] = 1.0f / (1.0f + expf(-v));
    }
  }
}

// ---------------- sparse aggregation: hm = h + sum_k att * h[src] ----------------
__global__ __launch_bounds__(192) void k_agg(
    const float* __restrict__ h, const float* __restrict__ eatt,
    const int* __restrict__ idx, _Float16* __restrict__ hm16) {
  int v = blockIdx.x, c = threadIdx.x;
  int b = v / N_;
  float acc = h[(size_t)v * C_ + c];
#pragma unroll
  for (int kk = 0; kk < K_; ++kk) {
    int s = b * N_ + idx[(size_t)v * K_ + kk];
    acc += eatt[(size_t)v * K_ + kk] * h[(size_t)s * C_ + c];
  }
  hm16[(size_t)v * C_ + c] = (_Float16)acc;
}

// ---------------- GIN update GEMM: relu((h+m) @ W + b) ----------------
__global__ __launch_bounds__(128) void k_gemm_relu(
    const _Float16* __restrict__ in16, const _Float16* __restrict__ wt,
    const float* __restrict__ bias, float* __restrict__ out32,
    _Float16* __restrict__ out16) {
  __shared__ __align__(16) _Float16 As[16][200];
  const int m0 = blockIdx.x * 16;
  const int tid = threadIdx.x, lane = tid & 31, wv = tid >> 5;
  for (int t = tid; t < 16 * C_; t += 128) {
    int row = t / C_, k = t - row * C_;
    As[row][k] = in16[(size_t)(m0 + row) * C_ + k];
  }
  __syncthreads();
  const _Float16* arow = &As[lane & 15][0];
  for (int nt = wv * 3; nt < wv * 3 + 3; ++nt) {
    v8f acc = {};
#pragma unroll
    for (int ks = 0; ks < 6; ++ks) {
      v16h a = load_a_frag(arow, lane, ks * 32);
      v16h bb = load_b_frag(wt, C_, lane, ks * 32, nt * 16);
      acc = wmma16(a, bb, acc);
    }
    int col = nt * 16 + (lane & 15);
    float bv = bias[col];
    int mb = (lane >> 4) << 3;
#pragma unroll
    for (int r = 0; r < 8; ++r) {
      float v = acc[r] + bv; v = v > 0.f ? v : 0.f;
      size_t o = (size_t)(m0 + r + mb) * C_ + col;
      out32[o] = v; out16[o] = (_Float16)v;
    }
  }
}

// ---------------- FFN: fused GEMM1+gelu (hidden in LDS) + GEMM2 + residual ----------
__global__ __launch_bounds__(128) void k_ffn(
    const _Float16* __restrict__ h16, const float* __restrict__ hres,
    const _Float16* __restrict__ w1t, const float* __restrict__ b1,
    const float* __restrict__ g1, const float* __restrict__ be1,
    const _Float16* __restrict__ w2t, const float* __restrict__ b2,
    const float* __restrict__ g2, const float* __restrict__ be2,
    float* __restrict__ tout) {
  __shared__ __align__(16) _Float16 As[16][200];
  __shared__ __align__(16) _Float16 Us[16][776];    // 16 x 768 hidden tile (+pad)
  const int m0 = blockIdx.x * 16;
  const int tid = threadIdx.x, lane = tid & 31, wv = tid >> 5;
  for (int t = tid; t < 16 * C_; t += 128) {
    int row = t / C_, k = t - row * C_;
    As[row][k] = h16[(size_t)(m0 + row) * C_ + k];
  }
  __syncthreads();
  const _Float16* arow = &As[lane & 15][0];
  for (int nt = wv * 12; nt < wv * 12 + 12; ++nt) { // 48 hidden N-tiles over 4 waves
    v8f acc = {};
#pragma unroll
    for (int ks = 0; ks < 6; ++ks) {
      v16h a = load_a_frag(arow, lane, ks * 32);
      v16h bb = load_b_frag(w1t, C_, lane, ks * 32, nt * 16);
      acc = wmma16(a, bb, acc);
    }
    int col = nt * 16 + (lane & 15);
    float bv = b1[col], gv = g1[col], bev = be1[col];
    int mb = (lane >> 4) << 3;
#pragma unroll
    for (int r = 0; r < 8; ++r) {
      float u = gelu_exact((acc[r] + bv) * gv + bev);
      Us[r + mb][col] = (_Float16)u;
    }
  }
  __syncthreads();
  const _Float16* urow = &Us[lane & 15][0];
  for (int nt = wv * 3; nt < wv * 3 + 3; ++nt) {
    v8f acc = {};
#pragma unroll
    for (int ks = 0; ks < 24; ++ks) {
      v16h a = load_a_frag(urow, lane, ks * 32);
      v16h bb = load_b_frag(w2t, HID_, lane, ks * 32, nt * 16);
      acc = wmma16(a, bb, acc);
    }
    int col = nt * 16 + (lane & 15);
    float bv = b2[col], gv = g2[col], bev = be2[col];
    int mb = (lane >> 4) << 3;
#pragma unroll
    for (int r = 0; r < 8; ++r) {
      size_t o = (size_t)(m0 + r + mb) * C_ + col;
      tout[o] = hres[o] + (acc[r] + bv) * gv + bev;
    }
  }
}

// ---------------- pooling + prediction head (tiny, VALU) ----------------
__global__ __launch_bounds__(192) void k_pool(const float* __restrict__ t,
                                              float* __restrict__ g) {
  int b = blockIdx.x, c = threadIdx.x;
  float s = 0.f;
  for (int n = 0; n < N_; ++n) s += t[((size_t)b * N_ + n) * C_ + c];
  g[b * C_ + c] = s * (1.0f / N_);
}

__global__ __launch_bounds__(256) void k_head(
    const float* __restrict__ g, const float* __restrict__ w1,
    const float* __restrict__ b1, const float* __restrict__ gm,
    const float* __restrict__ be, const float* __restrict__ w2,
    const float* __restrict__ b2, float* __restrict__ out) {
  __shared__ float red[256];
  int b = blockIdx.x, tid = threadIdx.x;
  float s = 0.f;
  for (int j = tid; j < PHID_; j += 256) {
    float d = 0.f;
    for (int c = 0; c < C_; ++c) d += g[b * C_ + c] * w1[(size_t)c * PHID_ + j];
    s += gelu_exact((d + b1[j]) * gm[j] + be[j]) * w2[j];
  }
  red[tid] = s;
  __syncthreads();
  for (int off = 128; off > 0; off >>= 1) {
    if (tid < off) red[tid] += red[tid + off];
    __syncthreads();
  }
  if (tid == 0) out[b] = red[0] + b2[0];
}

// ---------------- launch ----------------
extern "C" void kernel_launch(void* const* d_in, const int* in_sizes, int n_in,
                              void* d_out, int out_size, void* d_ws, size_t ws_size,
                              hipStream_t stream) {
  (void)in_sizes; (void)n_in; (void)out_size; (void)ws_size;
  const float* x       = (const float*)d_in[0];
  const float* stem_w  = (const float*)d_in[1];
  const float* stem_b  = (const float*)d_in[2];
  const float* pos     = (const float*)d_in[3];
  const float* att_w1  = (const float*)d_in[4];
  const float* att_b1  = (const float*)d_in[5];
  const float* att_w2  = (const float*)d_in[6];
  const float* att_b2  = (const float*)d_in[7];
  const float* gnn_w1  = (const float*)d_in[8];
  const float* gnn_b1  = (const float*)d_in[9];
  const float* gnn_w2  = (const float*)d_in[10];
  const float* gnn_b2  = (const float*)d_in[11];
  const float* ffn_w1  = (const float*)d_in[12];
  const float* ffn_b1  = (const float*)d_in[13];
  const float* ffn_g1  = (const float*)d_in[14];
  const float* ffn_be1 = (const float*)d_in[15];
  const float* ffn_w2  = (const float*)d_in[16];
  const float* ffn_b2  = (const float*)d_in[17];
  const float* ffn_g2  = (const float*)d_in[18];
  const float* ffn_be2 = (const float*)d_in[19];
  const float* pred_w1 = (const float*)d_in[20];
  const float* pred_b1 = (const float*)d_in[21];
  const float* pred_g  = (const float*)d_in[22];
  const float* pred_be = (const float*)d_in[23];
  const float* pred_w2 = (const float*)d_in[24];
  const float* pred_b2 = (const float*)d_in[25];

  float* eatt = (float*)d_out;           // first output: edge_att [E]
  float* pred = (float*)d_out + E_;      // second output: prediction [B]

  char* w = (char*)d_ws;
  auto take = [&](size_t bytes) {
    char* p = w; w += (bytes + 255) & ~(size_t)255; return p;
  };
  float*    nf    = (float*)take((size_t)BN_ * C_ * 4);     // h0 / h2 fp32
  float*    hbuf  = (float*)take((size_t)BN_ * C_ * 4);     // h1 fp32
  float*    tbuf  = (float*)take((size_t)BN_ * C_ * 4);     // FFN output
  _Float16* h16a  = (_Float16*)take((size_t)BN_ * C_ * 2);
  _Float16* h16b  = (_Float16*)take((size_t)BN_ * C_ * 2);
  _Float16* hm16  = (_Float16*)take((size_t)BN_ * C_ * 2);
  float*    sq    = (float*)take((size_t)BN_ * 4);
  int*      idxb  = (int*)take((size_t)BN_ * K_ * 4);
  float*    gpool = (float*)take((size_t)B_ * C_ * 4);
  _Float16* wstem = (_Float16*)take(768 * 192 * 2);
  _Float16* watt1 = (_Float16*)take(384 * 192 * 2);
  _Float16* wg1   = (_Float16*)take(192 * 192 * 2);
  _Float16* wg2   = (_Float16*)take(192 * 192 * 2);
  _Float16* wf1   = (_Float16*)take(192 * 768 * 2);
  _Float16* wf2   = (_Float16*)take(768 * 192 * 2);

  auto cg = [](int n) { return (n + 255) / 256; };
  // stem_w is OIHW = already n-major [192][768]; others need k->n transpose
  k_cvt <<<cg(768 * 192), 256, 0, stream>>>(stem_w, wstem, 768 * 192);
  k_cvtT<<<cg(384 * 192), 256, 0, stream>>>(att_w1, watt1, 384, 192);
  k_cvtT<<<cg(192 * 192), 256, 0, stream>>>(gnn_w1, wg1, 192, 192);
  k_cvtT<<<cg(192 * 192), 256, 0, stream>>>(gnn_w2, wg2, 192, 192);
  k_cvtT<<<cg(192 * 768), 256, 0, stream>>>(ffn_w1, wf1, 192, 768);
  k_cvtT<<<cg(768 * 192), 256, 0, stream>>>(ffn_w2, wf2, 768, 192);

  k_stem<<<dim3(36, B_), 128, 0, stream>>>(x, wstem, stem_b, pos, nf, h16a);
  k_sq<<<(BN_ + 255) / 256, 256, 0, stream>>>(nf, sq);
  k_knn<<<dim3(36, B_), 128, 0, stream>>>(h16a, sq, idxb);
  k_edge<<<E_ / 16, 32, 0, stream>>>(h16a, idxb, watt1, att_b1, att_w2, att_b2, eatt);

  // GIN layer 1
  k_agg<<<BN_, C_, 0, stream>>>(nf, eatt, idxb, hm16);
  k_gemm_relu<<<BN_ / 16, 128, 0, stream>>>(hm16, wg1, gnn_b1, hbuf, h16b);
  // GIN layer 2 (h0 dead -> reuse nf)
  k_agg<<<BN_, C_, 0, stream>>>(hbuf, eatt, idxb, hm16);
  k_gemm_relu<<<BN_ / 16, 128, 0, stream>>>(hm16, wg2, gnn_b2, nf, h16a);

  // FFN with residual
  k_ffn<<<BN_ / 16, 128, 0, stream>>>(h16a, nf, wf1, ffn_b1, ffn_g1, ffn_be1,
                                      wf2, ffn_b2, ffn_g2, ffn_be2, tbuf);

  k_pool<<<B_, C_, 0, stream>>>(tbuf, gpool);
  k_head<<<B_, 256, 0, stream>>>(gpool, pred_w1, pred_b1, pred_g, pred_be,
                                 pred_w2, pred_b2, pred);
}
